// AttentionBase_78039555768786
// MI455X (gfx1250) — matrix operand order repaired and checked by reading, not compiled, not run on hardware
//
#include <hip/hip_runtime.h>

typedef _Float16 v16h __attribute__((ext_vector_type(16)));
typedef _Float16 v8h  __attribute__((ext_vector_type(8)));
typedef _Float16 v4h  __attribute__((ext_vector_type(4)));
typedef float    v8f  __attribute__((ext_vector_type(8)));
typedef float    v4f  __attribute__((ext_vector_type(4)));

#define BATCH 16
#define LSEQ  1024
#define DDIM  512
#define KC    64      // keys per chunk
#define MT    32      // q rows per workgroup (2 row tiles of 16)

// LDS layout (units: _Float16; all strides multiples of 8 halfs = 16B)
#define KS_STRIDE 520
#define KT_STRIDE 72
#define P_STRIDE  72
#define KS_OFF 0
#define KT_OFF (KC * KS_STRIDE)                  // 33280
#define P_OFF  (KT_OFF + DDIM * KT_STRIDE)       // 70144
#define HALFS_TOTAL (P_OFF + 2 * 16 * P_STRIDE)  // 72448
#define RS_OFF_B   (HALFS_TOTAL * 2)             // row-sum region (bytes)
#define STAT_OFF_B (RS_OFF_B + 2 * 4 * 16 * 4)   // mean/var region
#define SMEM_BYTES (STAT_OFF_B + 2 * 4 * 16 * 2 * 4)   // 146432 B -> 2 WGs / WGP

#define SHUF16(lo, hi) __builtin_shufflevector(lo, hi, 0,1,2,3,4,5,6,7,8,9,10,11,12,13,14,15)

// scheduling pins: 0x100 = DS read group, 0x008 = MFMA/WMMA group
#define SGB(mask, size) __builtin_amdgcn_sched_group_barrier((mask), (size), 0)

__global__ __launch_bounds__(256) void
attn_tanh_ln_kernel(const float* __restrict__ query,
                    const float* __restrict__ context,
                    const float* __restrict__ gamma,
                    const float* __restrict__ beta,
                    float* __restrict__ out)
{
    extern __shared__ __align__(16) char smem_raw[];
    _Float16* sh   = (_Float16*)smem_raw;
    float*    rsum = (float*)(smem_raw + RS_OFF_B);    // [2][4][16]
    float*    stat = (float*)(smem_raw + STAT_OFF_B);  // [2][4][16][2]

    const int b     = blockIdx.y;
    const int qbase = blockIdx.x * MT;
    const int tid   = threadIdx.x;
    const int lane  = tid & 31;
    const int wave  = tid >> 5;
    const int wr    = wave >> 2;   // row-tile 0/1
    const int wc    = wave & 3;    // col group (128 out cols) / key sub-tile
    const int n16   = lane & 15;
    const int hsel  = lane >> 4;   // half-wave select

    // ---- preload all Q A-fragments (16 slices of K=32) straight from global ----
    // A layout: lane M = lane%16; halfs 0..7 = d0+hsel*8..+7, halfs 8..15 = d0+16+hsel*8..+7
    v16h qa[16];
    {
        const float* qrowg = query + ((size_t)b * LSEQ + qbase + wr * 16 + n16) * DDIM;
        #pragma unroll
        for (int s16 = 0; s16 < 16; ++s16) {
            const int d0 = s16 * 32;
            v4f f0 = *(const v4f*)(qrowg + d0 + hsel * 8);
            v4f f1 = *(const v4f*)(qrowg + d0 + hsel * 8 + 4);
            v4f f2 = *(const v4f*)(qrowg + d0 + 16 + hsel * 8);
            v4f f3 = *(const v4f*)(qrowg + d0 + 16 + hsel * 8 + 4);
            v16h a;
            a[0]  = (_Float16)f0[0]; a[1]  = (_Float16)f0[1];
            a[2]  = (_Float16)f0[2]; a[3]  = (_Float16)f0[3];
            a[4]  = (_Float16)f1[0]; a[5]  = (_Float16)f1[1];
            a[6]  = (_Float16)f1[2]; a[7]  = (_Float16)f1[3];
            a[8]  = (_Float16)f2[0]; a[9]  = (_Float16)f2[1];
            a[10] = (_Float16)f2[2]; a[11] = (_Float16)f2[3];
            a[12] = (_Float16)f3[0]; a[13] = (_Float16)f3[1];
            a[14] = (_Float16)f3[2]; a[15] = (_Float16)f3[3];
            qa[s16] = a;
        }
    }

    const v8f zero8 = {0.f,0.f,0.f,0.f,0.f,0.f,0.f,0.f};
    v8f acc[8];
    #pragma unroll
    for (int t = 0; t < 8; ++t) acc[t] = zero8;
    float rs[8];
    #pragma unroll
    for (int j = 0; j < 8; ++j) rs[j] = 0.f;

    const float scale = 0.044194173824159216f;   // 1/sqrt(512)
    const float L2E   = 1.4426950408889634f;

    #pragma unroll 1
    for (int kc = 0; kc < LSEQ / KC; ++kc) {
        __syncthreads();   // previous chunk's GEMM2 done reading Ks/Kt/P

        // ---- stage context chunk (64 keys x 512) f32->f16 in two layouts ----
        const float* csrc = context + ((size_t)b * LSEQ + kc * KC) * DDIM;
        if (kc + 1 < LSEQ / KC) {                 // prefetch next chunk
            const float* nxt = csrc + (size_t)KC * DDIM;
            #pragma unroll
            for (int i = 0; i < 4; ++i)
                __builtin_prefetch(nxt + tid * 128 + i * 32, 0, 1);
        }
        #pragma unroll 8
        for (int i = 0; i < 32; ++i) {
            int v   = tid + i * 256;      // float4 index over 64*128
            int key = v >> 7;
            int c4  = (v & 127) << 2;
            v4f f = *(const v4f*)(csrc + key * DDIM + c4);
            v4h h = { (_Float16)f[0], (_Float16)f[1], (_Float16)f[2], (_Float16)f[3] };
            *(v4h*)(sh + KS_OFF + key * KS_STRIDE + c4) = h;   // row-major [key][d]
            _Float16* kt = sh + KT_OFF + key;                  // transposed [d][key]
            kt[(c4 + 0) * KT_STRIDE] = h[0];
            kt[(c4 + 1) * KT_STRIDE] = h[1];
            kt[(c4 + 2) * KT_STRIDE] = h[2];
            kt[(c4 + 3) * KT_STRIDE] = h[3];
        }
        __syncthreads();

        // ---- GEMM1: S(16x16) = Q(16xD) . K^T for keys [wc*16, +16) ----
        // 4-slot rotating B buffers + sched_group_barrier pins give 2-pair lookahead:
        //   [6 DS] W [2 DS] W [2 DS] ... -> partial s_wait_dscnt instead of full drains
        const int myk = wc * 16;
        const _Float16* krow = sh + KS_OFF + (myk + n16) * KS_STRIDE;  // B: N = lane%16
        v8f s = zero8;
        v8h blo[4], bhi[4];
        blo[0] = *(const v8h*)(krow + hsel * 8);
        bhi[0] = *(const v8h*)(krow + 16 + hsel * 8);
        #pragma unroll
        for (int s16 = 0; s16 < 16; ++s16) {
            const int cur = s16 & 3, nxt = (s16 + 1) & 3;
            if (s16 < 15) {
                blo[nxt] = *(const v8h*)(krow + (s16 + 1) * 32 + hsel * 8);
                bhi[nxt] = *(const v8h*)(krow + (s16 + 1) * 32 + 16 + hsel * 8);
            }
            v16h bm = SHUF16(blo[cur], bhi[cur]);
            s = __builtin_amdgcn_wmma_f32_16x16x32_f16(false, qa[s16], false, bm,
                                                       (short)0, s, false, false);
            if (s16 == 0)            SGB(0x100, 6);
            else if (s16 <= 13)      SGB(0x100, 2);
            SGB(0x008, 1);
        }

        // ---- p = exp(tanh(s*scale)); row sums; share P via LDS as f16 ----
        _Float16* pb = sh + P_OFF + wr * (16 * P_STRIDE);
        #pragma unroll
        for (int j = 0; j < 8; ++j) {
            float x  = s[j] * scale;
            float e  = __builtin_amdgcn_exp2f(x * (2.0f * L2E));           // e^(2x)
            float th = (e - 1.0f) * __builtin_amdgcn_rcpf(e + 1.0f);       // tanh(x)
            float p  = __builtin_amdgcn_exp2f(th * L2E);                   // e^tanh
            float red = p;                                                 // 16-lane half sum
            red += __shfl_xor(red, 1, 32);
            red += __shfl_xor(red, 2, 32);
            red += __shfl_xor(red, 4, 32);
            red += __shfl_xor(red, 8, 32);
            rs[j] += red;
            pb[(j + hsel * 8) * P_STRIDE + myk + n16] = (_Float16)p;       // C layout -> row major
        }
        __syncthreads();

        // ---- GEMM2: acc(16x128) += P(16x64) . V(64x128), pipelined like GEMM1 ----
        const _Float16* prow = sh + P_OFF + wr * (16 * P_STRIDE) + n16 * P_STRIDE;
        v16h pa[2];
        #pragma unroll
        for (int ks = 0; ks < 2; ++ks) {
            v8h lo = *(const v8h*)(prow + ks * 32 + hsel * 8);
            v8h hi = *(const v8h*)(prow + ks * 32 + 16 + hsel * 8);
            pa[ks] = SHUF16(lo, hi);
        }
        const _Float16* vbase = sh + KT_OFF + (wc * 128 + n16) * KT_STRIDE;
        v8h wlo[4], whi[4];
        wlo[0] = *(const v8h*)(vbase + hsel * 8);
        whi[0] = *(const v8h*)(vbase + 16 + hsel * 8);
        #pragma unroll
        for (int u = 0; u < 16; ++u) {           // u = t*2 + ks
            const int cur = u & 3, nxt = (u + 1) & 3;
            if (u < 15) {
                const int un = u + 1;
                const _Float16* vb = vbase + (un >> 1) * (16 * KT_STRIDE) + (un & 1) * 32;
                wlo[nxt] = *(const v8h*)(vb + hsel * 8);
                whi[nxt] = *(const v8h*)(vb + 16 + hsel * 8);
            }
            v16h bm = SHUF16(wlo[cur], whi[cur]);
            acc[u >> 1] = __builtin_amdgcn_wmma_f32_16x16x32_f16(false, pa[u & 1], false, bm,
                                                                 (short)0, acc[u >> 1], false, false);
            if (u == 0)              SGB(0x100, 10);   // 4 pa loads + 3 B pairs
            else if (u <= 13)        SGB(0x100, 2);
            SGB(0x008, 1);
        }
    }

    // ---- combine row sums across the 4 col-waves ----
    if (n16 == 0) {
        float* dst = rsum + (wr * 4 + wc) * 16 + hsel * 8;
        #pragma unroll
        for (int j = 0; j < 8; ++j) dst[j] = rs[j];
    }
    __syncthreads();
    float linv[8];
    #pragma unroll
    for (int j = 0; j < 8; ++j) {
        int m = j + hsel * 8;
        float l = rsum[(wr * 4 + 0) * 16 + m] + rsum[(wr * 4 + 1) * 16 + m]
                + rsum[(wr * 4 + 2) * 16 + m] + rsum[(wr * 4 + 3) * 16 + m];
        linv[j] = __builtin_amdgcn_rcpf(l);
    }

    // ---- x = acc/l + q (fp32 residual); per-wave partial LN stats ----
    const int cbase = wc * 128;
    float psum[8], psq[8];
    #pragma unroll
    for (int j = 0; j < 8; ++j) { psum[j] = 0.f; psq[j] = 0.f; }
    #pragma unroll
    for (int t = 0; t < 8; ++t) {
        int col = cbase + t * 16 + n16;
        #pragma unroll
        for (int j = 0; j < 8; ++j) {
            int qrow = qbase + wr * 16 + j + hsel * 8;
            float q = query[((size_t)b * LSEQ + qrow) * DDIM + col];
            float x = acc[t][j] * linv[j] + q;
            acc[t][j] = x;
            psum[j] += x;
            psq[j]  += x * x;
        }
    }
    #pragma unroll
    for (int j = 0; j < 8; ++j) {
        float a1 = psum[j], a2 = psq[j];
        a1 += __shfl_xor(a1, 1, 32);  a2 += __shfl_xor(a2, 1, 32);
        a1 += __shfl_xor(a1, 2, 32);  a2 += __shfl_xor(a2, 2, 32);
        a1 += __shfl_xor(a1, 4, 32);  a2 += __shfl_xor(a2, 4, 32);
        a1 += __shfl_xor(a1, 8, 32);  a2 += __shfl_xor(a2, 8, 32);
        if (n16 == 0) {
            int m = j + hsel * 8;
            stat[(((wr * 4 + wc) * 16) + m) * 2 + 0] = a1;
            stat[(((wr * 4 + wc) * 16) + m) * 2 + 1] = a2;
        }
    }
    __syncthreads();
    float mu[8], inv[8];
    #pragma unroll
    for (int j = 0; j < 8; ++j) {
        int m = j + hsel * 8;
        float s1 = 0.f, s2 = 0.f;
        #pragma unroll
        for (int w2 = 0; w2 < 4; ++w2) {
            s1 += stat[(((wr * 4 + w2) * 16) + m) * 2 + 0];
            s2 += stat[(((wr * 4 + w2) * 16) + m) * 2 + 1];
        }
        float mean = s1 * (1.0f / DDIM);
        float var  = s2 * (1.0f / DDIM) - mean * mean;
        mu[j]  = mean;
        inv[j] = __builtin_amdgcn_rsqf(var + 1e-6f);
    }
    #pragma unroll
    for (int t = 0; t < 8; ++t) {
        int col = cbase + t * 16 + n16;
        float g  = gamma[col];
        float be = beta[col];
        #pragma unroll
        for (int j = 0; j < 8; ++j) {
            int qrow = qbase + wr * 16 + j + hsel * 8;
            out[((size_t)b * LSEQ + qrow) * DDIM + col] =
                g * (acc[t][j] - mu[j]) * inv[j] + be;
        }
    }
}

extern "C" void kernel_launch(void* const* d_in, const int* in_sizes, int n_in,
                              void* d_out, int out_size, void* d_ws, size_t ws_size,
                              hipStream_t stream) {
    (void)in_sizes; (void)n_in; (void)out_size; (void)d_ws; (void)ws_size;
    const float* query   = (const float*)d_in[0];
    const float* context = (const float*)d_in[1];
    const float* gamma   = (const float*)d_in[2];
    const float* beta    = (const float*)d_in[3];
    float* out = (float*)d_out;

    dim3 grid(LSEQ / MT, BATCH);   // (32, 16)
    dim3 block(256);               // 8 wave32
    attn_tanh_ln_kernel<<<grid, block, SMEM_BYTES, stream>>>(query, context, gamma, beta, out);
}
